// RobertaSelfAttention_39513699123651
// MI455X (gfx1250) — compile-verified, matrix operand order checked by reading, not run on hardware
//
#include <hip/hip_runtime.h>

typedef __attribute__((ext_vector_type(16))) __bf16 bf16x16;
typedef __attribute__((ext_vector_type(8)))  float  f32x8;
typedef __attribute__((ext_vector_type(4)))  unsigned int u32x4;
typedef __attribute__((ext_vector_type(8)))  int i32x8;
typedef __attribute__((ext_vector_type(4)))  int i32x4;

static constexpr int NHEAD = 16;
static constexpr int HDIM  = 64;
static constexpr int SEQ   = 1024;
static constexpr int HID   = 1024;

static __device__ inline unsigned short f32_to_bf16_bits(float f) {
    unsigned int u = __float_as_uint(f);
    unsigned int r = (u + 0x7FFFu + ((u >> 16) & 1u)) >> 16;
    return (unsigned short)r;
}

// ---------------------------------------------------------------------------
// TDM: 2D tile load Global -> LDS, data_size = 4B units, optional row padding.
// Descriptor packing per CDNA5 ISA section 8 (D# group0/group1).
// ---------------------------------------------------------------------------
static __device__ inline void tdm_load_2d(unsigned lds_off, const void* gaddr,
                                          unsigned tensor_d0, unsigned tensor_d1,
                                          unsigned tile_d0, unsigned tile_d1,
                                          unsigned d0_stride,
                                          unsigned pad_interval, unsigned pad_amount) {
    unsigned long long ga = (unsigned long long)(size_t)gaddr;
    u32x4 g0;
    g0[0] = 1u;                                        // count=1 (valid), no gather
    g0[1] = lds_off;                                   // LDS byte address
    g0[2] = (unsigned)(ga & 0xFFFFFFFFu);              // global_addr[31:0]
    g0[3] = (unsigned)((ga >> 32) & 0x01FFFFFFu)       // global_addr[56:32]
          | (2u << 30);                                // type = 2 ("image")
    i32x8 g1;
    g1[0] = (int)((2u << 16)                           // data_size = 4 bytes
          | (1u << 20)                                 // pad_enable
          | (pad_interval << 22) | (pad_amount << 25));
    g1[1] = (int)((tensor_d0 & 0xFFFFu) << 16);        // tensor_dim0[15:0]
    g1[2] = (int)(((tensor_d0 >> 16) & 0xFFFFu) | ((tensor_d1 & 0xFFFFu) << 16));
    g1[3] = (int)(((tensor_d1 >> 16) & 0xFFFFu) | ((tile_d0 & 0xFFFFu) << 16));
    g1[4] = (int)(tile_d1 & 0xFFFFu);                  // tile_dim1, tile_dim2 = 0
    g1[5] = (int)d0_stride;                            // tensor_dim0_stride[31:0]
    g1[6] = 0;                                         // stride hi / dim1_stride lo
    g1[7] = 0;
    i32x4 z4 = {0, 0, 0, 0};
#if defined(__clang_major__) && (__clang_major__ >= 23)
    i32x8 z8 = {0, 0, 0, 0, 0, 0, 0, 0};
    __builtin_amdgcn_tensor_load_to_lds(g0, g1, z4, z4, z8, 0);
#else
    __builtin_amdgcn_tensor_load_to_lds(g0, g1, z4, z4, 0);
#endif
}

// ---------------------------------------------------------------------------
// Stage 1a: fp32 -> bf16 elementwise (hidden states)
// ---------------------------------------------------------------------------
__global__ void cvt_f32_bf16(const float* __restrict__ in,
                             unsigned short* __restrict__ out, int n) {
    int i = (blockIdx.x * blockDim.x + threadIdx.x) * 4;
    if (i + 3 < n) {
        float4 f = *(const float4*)(in + i);
        unsigned int lo = (unsigned int)f32_to_bf16_bits(f.x) |
                          ((unsigned int)f32_to_bf16_bits(f.y) << 16);
        unsigned int hi = (unsigned int)f32_to_bf16_bits(f.z) |
                          ((unsigned int)f32_to_bf16_bits(f.w) << 16);
        *(uint2*)(out + i) = make_uint2(lo, hi);
    }
}

// ---------------------------------------------------------------------------
// Stage 1b: fp32 W[k][n] -> bf16 Wt[n][k] (one-time tiled transpose, 64x64)
// ---------------------------------------------------------------------------
__global__ __launch_bounds__(256)
void cvt_transpose(const float* __restrict__ W, unsigned short* __restrict__ Wt) {
    __shared__ unsigned short Ls[64 * 72];
    const int tid = threadIdx.x;
    const int n0 = blockIdx.x * 64;
    const int k0 = blockIdx.y * 64;
    #pragma unroll
    for (int it = 0; it < 4; ++it) {
        int c = tid + it * 256;              // 1024 chunks of 4 floats
        int row = c >> 4, col4 = (c & 15) * 4;
        float4 f = *(const float4*)(&W[(size_t)(k0 + row) * HID + n0 + col4]);
        Ls[(col4 + 0) * 72 + row] = f32_to_bf16_bits(f.x);
        Ls[(col4 + 1) * 72 + row] = f32_to_bf16_bits(f.y);
        Ls[(col4 + 2) * 72 + row] = f32_to_bf16_bits(f.z);
        Ls[(col4 + 3) * 72 + row] = f32_to_bf16_bits(f.w);
    }
    __syncthreads();
    #pragma unroll
    for (int it = 0; it < 2; ++it) {
        int c = tid + it * 256;              // 512 chunks of 8 bf16
        int nrow = c >> 3, kcol = (c & 7) * 8;
        *(uint4*)(&Wt[(size_t)(n0 + nrow) * HID + k0 + kcol]) =
            *(const uint4*)(&Ls[nrow * 72 + kcol]);
    }
}

// ---------------------------------------------------------------------------
// Stage 2: QKV GEMM. C[4096,1024] = X * W + b.  W pre-transposed (Wt[n][k]).
// Block tile 128x128, 8 waves, wave tile 32x64 (2x4 WMMA => 8 wmma / k-step).
// transposeOut=1 stores [B][NH][HD][S] (for V), else [B][NH][S][HD].
// ---------------------------------------------------------------------------
static constexpr int APITCH = 40;
static constexpr int BPITCH = 40;

__global__ __launch_bounds__(256)
void qkv_gemm(const unsigned short* __restrict__ X,
              const unsigned short* __restrict__ Wt,
              const float* __restrict__ bias,
              unsigned short* __restrict__ out,
              float oscale, int transposeOut)
{
    __shared__ unsigned short As[128 * APITCH];   // [m][k]
    __shared__ unsigned short Bs[128 * BPITCH];   // [n][k]

    const int tid  = threadIdx.x;
    const int lane = tid & 31;
    const int w    = tid >> 5;
    const int wr   = w & 3;
    const int wc   = w >> 2;
    const int half = lane >> 4;
    const int l16  = lane & 15;
    const int n0   = blockIdx.x * 128;
    const int m0   = blockIdx.y * 128;

    f32x8 acc[2][4] = {};

    for (int kk = 0; kk < HID; kk += 32) {
        #pragma unroll
        for (int it = 0; it < 2; ++it) {      // A: 128x32, 512 x 16B chunks
            int c = tid + it * 256;
            int row = c >> 2, col = (c & 3) * 8;
            *(uint4*)(&As[row * APITCH + col]) =
                *(const uint4*)(&X[(size_t)(m0 + row) * HID + kk + col]);
        }
        #pragma unroll
        for (int it = 0; it < 2; ++it) {      // B: 128x32 from Wt rows (no scatter)
            int c = tid + it * 256;
            int row = c >> 2, col = (c & 3) * 8;
            *(uint4*)(&Bs[row * BPITCH + col]) =
                *(const uint4*)(&Wt[(size_t)(n0 + row) * HID + kk + col]);
        }
        __syncthreads();

        union FR { bf16x16 v; uint4 q[2]; };
        FR a[2], bf[4];
        #pragma unroll
        for (int mi = 0; mi < 2; ++mi) {
            const unsigned short* rp = &As[(32 * wr + 16 * mi + l16) * APITCH];
            a[mi].q[0] = *(const uint4*)(rp + 8 * half);
            a[mi].q[1] = *(const uint4*)(rp + 16 + 8 * half);
        }
        #pragma unroll
        for (int ni = 0; ni < 4; ++ni) {
            const unsigned short* rp = &Bs[(64 * wc + 16 * ni + l16) * BPITCH];
            bf[ni].q[0] = *(const uint4*)(rp + 8 * half);
            bf[ni].q[1] = *(const uint4*)(rp + 16 + 8 * half);
        }
        #pragma unroll
        for (int mi = 0; mi < 2; ++mi)
            #pragma unroll
            for (int ni = 0; ni < 4; ++ni)
                acc[mi][ni] = __builtin_amdgcn_wmma_f32_16x16x32_bf16(
                    false, a[mi].v, false, bf[ni].v,
                    (short)0, acc[mi][ni], false, false);
        __syncthreads();
    }

    #pragma unroll
    for (int mi = 0; mi < 2; ++mi) {
        #pragma unroll
        for (int ni = 0; ni < 4; ++ni) {
            int ncol = n0 + 64 * wc + 16 * ni + l16;
            float bv = bias[ncol];
            int hh = ncol >> 6, dd = ncol & 63;
            #pragma unroll
            for (int r = 0; r < 8; ++r) {
                int mrow = m0 + 32 * wr + 16 * mi + r + 8 * half;
                int bb = mrow >> 10, ss = mrow & 1023;
                float v = (acc[mi][ni][r] + bv) * oscale;
                size_t base = (size_t)(bb * NHEAD + hh);
                size_t idx = transposeOut ? (base * HDIM + dd) * SEQ + ss
                                          : (base * SEQ + ss) * HDIM + dd;
                out[idx] = f32_to_bf16_bits(v);
            }
        }
    }
}

// ---------------------------------------------------------------------------
// Stage 3: flash attention. 8 waves x 16 q-rows = 128 q rows per block.
// K tile [32 key][64 hd] and V tile [64 hd][32 key] (from pre-transposed Vt)
// staged by the Tensor Data Mover into double-buffered LDS (TDM padding
// reproduces the 72 / 40 element pitches); DMA of tile j+32 overlaps compute.
// ---------------------------------------------------------------------------
static constexpr int KP = 72;   // 64 + 8 pad (TDM: 32 dwords + 4 dwords pad)
static constexpr int VP = 40;   // 32 + 8 pad (TDM: 16 dwords + 4 dwords pad)
static constexpr int PP = 40;

__global__ __launch_bounds__(256)
void attn(const unsigned short* __restrict__ Q,
          const unsigned short* __restrict__ K,
          const unsigned short* __restrict__ Vt,
          const float* __restrict__ mask,
          float* __restrict__ out)
{
    __shared__ unsigned short Ks[2][32 * KP];     // [key][hd]
    __shared__ unsigned short Vs[2][64 * VP];     // [hd][key]
    __shared__ unsigned short Ps[8 * 16 * PP];

    const int tid  = threadIdx.x;
    const int lane = tid & 31;
    const int w    = tid >> 5;
    const int half = lane >> 4;
    const int l16  = lane & 15;
    const int q0   = blockIdx.x * 128;
    const int h    = blockIdx.y;
    const int b    = blockIdx.z;
    const size_t hb  = (size_t)(b * NHEAD + h) * SEQ * HDIM;   // Q,K base
    const size_t hbT = (size_t)(b * NHEAD + h) * HDIM * SEQ;   // Vt base

    union FR { bf16x16 v; uint4 q[2]; };

    FR qf[2];
    {
        const unsigned short* qp = &Q[hb + (size_t)(q0 + w * 16 + l16) * HDIM];
        #pragma unroll
        for (int ks = 0; ks < 2; ++ks) {
            qf[ks].q[0] = *(const uint4*)(qp + ks * 32 + 8 * half);
            qf[ks].q[1] = *(const uint4*)(qp + ks * 32 + 16 + 8 * half);
        }
    }

    f32x8 o[4] = {};
    float mrow[8], lrow[8];
    #pragma unroll
    for (int r = 0; r < 8; ++r) { mrow[r] = -3.0e38f; lrow[r] = 0.0f; }

    unsigned short* Pw = &Ps[w * 16 * PP];

    // prologue: DMA tile j=0 into buffer 0
    if (w == 0) {
        tdm_load_2d((unsigned)(size_t)&Ks[0][0], &K[hb],
                    /*t_d0=*/32, /*t_d1=*/SEQ, /*tile=*/32, 32,
                    /*stride=*/32, /*pad_int=*/4, /*pad_amt=*/3);
        tdm_load_2d((unsigned)(size_t)&Vs[0][0], &Vt[hbT],
                    /*t_d0=*/SEQ / 2, /*t_d1=*/HDIM, /*tile=*/16, 64,
                    /*stride=*/SEQ / 2, /*pad_int=*/3, /*pad_amt=*/3);
        __builtin_amdgcn_s_wait_tensorcnt(0);
    }
    __syncthreads();

    for (int j = 0; j < SEQ; j += 32) {
        const int buf = (j >> 5) & 1;
        const bool more = (j + 32) < SEQ;
        if (w == 0 && more) {   // async DMA of next tile overlaps this tile's math
            tdm_load_2d((unsigned)(size_t)&Ks[buf ^ 1][0],
                        &K[hb + (size_t)(j + 32) * HDIM],
                        32, SEQ, 32, 32, 32, 4, 3);
            tdm_load_2d((unsigned)(size_t)&Vs[buf ^ 1][0],
                        &Vt[hbT + (size_t)(j + 32)],
                        SEQ / 2, HDIM, 16, 64, SEQ / 2, 3, 3);
        }

        // scores = Q * K^T (scale prefolded into Q), additive mask
        f32x8 acc[2] = {};
        #pragma unroll
        for (int nt = 0; nt < 2; ++nt) {
            const unsigned short* kp = &Ks[buf][(nt * 16 + l16) * KP];
            #pragma unroll
            for (int ks = 0; ks < 2; ++ks) {
                FR kf;
                kf.q[0] = *(const uint4*)(kp + ks * 32 + 8 * half);
                kf.q[1] = *(const uint4*)(kp + ks * 32 + 16 + 8 * half);
                acc[nt] = __builtin_amdgcn_wmma_f32_16x16x32_bf16(
                    false, qf[ks].v, false, kf.v, (short)0, acc[nt], false, false);
            }
            float mv = mask[b * SEQ + j + nt * 16 + l16];
            #pragma unroll
            for (int r = 0; r < 8; ++r) acc[nt][r] += mv;
        }

        // online softmax (row = r + 8*half; 16-lane half-wave holds the cols)
        float p0[8], p1[8];
        #pragma unroll
        for (int r = 0; r < 8; ++r) {
            float tm = fmaxf(acc[0][r], acc[1][r]);
            #pragma unroll
            for (int off = 8; off >= 1; off >>= 1)
                tm = fmaxf(tm, __shfl_xor(tm, off, 16));
            float mnew  = fmaxf(mrow[r], tm);
            float scale = __expf(mrow[r] - mnew);
            float s0 = __expf(acc[0][r] - mnew);
            float s1 = __expf(acc[1][r] - mnew);
            p0[r] = s0; p1[r] = s1;
            float rs = s0 + s1;
            #pragma unroll
            for (int off = 8; off >= 1; off >>= 1)
                rs += __shfl_xor(rs, off, 16);
            lrow[r] = lrow[r] * scale + rs;
            mrow[r] = mnew;
            #pragma unroll
            for (int ot = 0; ot < 4; ++ot) o[ot][r] *= scale;
        }

        // P: C-layout regs -> per-wave LDS -> A-layout fragment
        #pragma unroll
        for (int r = 0; r < 8; ++r) {
            int prow = r + 8 * half;
            Pw[prow * PP + l16]      = f32_to_bf16_bits(p0[r]);
            Pw[prow * PP + 16 + l16] = f32_to_bf16_bits(p1[r]);
        }
        FR pf;
        {
            const unsigned short* pp = &Pw[l16 * PP];
            pf.q[0] = *(const uint4*)(pp + 8 * half);
            pf.q[1] = *(const uint4*)(pp + 16 + 8 * half);
        }

        // O += P(16x32) * V(32x64)
        #pragma unroll
        for (int ot = 0; ot < 4; ++ot) {
            const unsigned short* vp = &Vs[buf][(ot * 16 + l16) * VP];
            FR vf;
            vf.q[0] = *(const uint4*)(vp + 8 * half);
            vf.q[1] = *(const uint4*)(vp + 16 + 8 * half);
            o[ot] = __builtin_amdgcn_wmma_f32_16x16x32_bf16(
                false, pf.v, false, vf.v, (short)0, o[ot], false, false);
        }

        if (w == 0 && more) __builtin_amdgcn_s_wait_tensorcnt(0);
        __syncthreads();   // publish next buffer / protect P reuse
    }

    #pragma unroll
    for (int r = 0; r < 8; ++r) {
        float inv = 1.0f / lrow[r];
        int s = q0 + w * 16 + r + 8 * half;
        float* op = &out[((size_t)(b * SEQ + s)) * HID + h * HDIM];
        #pragma unroll
        for (int ot = 0; ot < 4; ++ot)
            op[ot * 16 + l16] = o[ot][r] * inv;
    }
}

// ---------------------------------------------------------------------------
extern "C" void kernel_launch(void* const* d_in, const int* in_sizes, int n_in,
                              void* d_out, int out_size, void* d_ws, size_t ws_size,
                              hipStream_t stream) {
    (void)in_sizes; (void)n_in; (void)out_size; (void)ws_size;
    const float* hs   = (const float*)d_in[0];
    const float* mask = (const float*)d_in[1];
    const float* Wq   = (const float*)d_in[2];
    const float* bq   = (const float*)d_in[3];
    const float* Wk   = (const float*)d_in[4];
    const float* bk   = (const float*)d_in[5];
    const float* Wv   = (const float*)d_in[6];
    const float* bv   = (const float*)d_in[7];
    float* out = (float*)d_out;

    char* ws = (char*)d_ws;
    unsigned short* hsB = (unsigned short*)(ws);
    unsigned short* WqT = (unsigned short*)(ws + ((size_t)8  << 20));
    unsigned short* WkT = (unsigned short*)(ws + ((size_t)10 << 20));
    unsigned short* WvT = (unsigned short*)(ws + ((size_t)12 << 20));
    unsigned short* Qb  = (unsigned short*)(ws + ((size_t)14 << 20));
    unsigned short* Kb  = (unsigned short*)(ws + ((size_t)22 << 20));
    unsigned short* Vb  = (unsigned short*)(ws + ((size_t)30 << 20));

    const int nHS = 4 * SEQ * HID;
    cvt_f32_bf16<<<(nHS / 4 + 255) / 256, 256, 0, stream>>>(hs, hsB, nHS);
    dim3 gt(HID / 64, HID / 64);   // (16,16)
    cvt_transpose<<<gt, 256, 0, stream>>>(Wq, WqT);
    cvt_transpose<<<gt, 256, 0, stream>>>(Wk, WkT);
    cvt_transpose<<<gt, 256, 0, stream>>>(Wv, WvT);

    dim3 gg(HID / 128, (4 * SEQ) / 128);   // (8, 32)
    qkv_gemm<<<gg, 256, 0, stream>>>(hsB, WqT, bq, Qb, 0.125f, 0);
    qkv_gemm<<<gg, 256, 0, stream>>>(hsB, WkT, bk, Kb, 1.0f, 0);
    qkv_gemm<<<gg, 256, 0, stream>>>(hsB, WvT, bv, Vb, 1.0f, 1);  // V transposed

    dim3 ga(SEQ / 128, NHEAD, 4);
    attn<<<ga, 256, 0, stream>>>(Qb, Kb, Vb, mask, out);
}
